// StreamingFlowVectorField_43714177139279
// MI455X (gfx1250) — matrix operand order repaired and checked by reading, not compiled, not run on hardware
//
#include <hip/hip_runtime.h>
#include <hip/hip_bf16.h>

typedef _Float16 f16_t;
typedef _Float16 v8h  __attribute__((ext_vector_type(8)));
typedef _Float16 v16h __attribute__((ext_vector_type(16)));
typedef float    v8f  __attribute__((ext_vector_type(8)));

#define AH       128   // ACTION_DIM * HORIZON
#define OBS_DIM  512
#define HIDDEN   256
#define BATCH    512
#define XDIM     640   // AH + OBS_DIM
#define H1PITCH  264   // 256 + 8 halves of padding (bank spread, keeps 16B align)

// Fast silu: v * v_rcp_f32(1 + e^-v).  v_rcp is ~1ulp; plenty for this net.
__device__ __forceinline__ float silu_f(float v) {
    return v * __builtin_amdgcn_rcpf(1.0f + __expf(-v));
}

// ---------------------------------------------------------------------------
// Prep: W2T_f16[n*256 + k] = (f16) W2[k*256 + n]   (column-major f16 copy)
// ---------------------------------------------------------------------------
__global__ void w2t_kernel(const float* __restrict__ W2, f16_t* __restrict__ w2t) {
    const int n = blockIdx.x;
    const int k = threadIdx.x;
    w2t[n * HIDDEN + k] = (f16_t)W2[k * HIDDEN + n];
}

// ---------------------------------------------------------------------------
// base[b,n] = b1[n] + t[b]*W1[2,n] + sum_k obs[b,k] * W1[3+k,n]   (fp32)
// ---------------------------------------------------------------------------
__global__ void base_kernel(const float* __restrict__ t, const float* __restrict__ x,
                            const float* __restrict__ W1, const float* __restrict__ b1,
                            float* __restrict__ base) {
    __shared__ float obs_s[OBS_DIM];
    const int b = blockIdx.x;
    const int n = threadIdx.x;
    obs_s[n]       = x[b * XDIM + AH + n];
    obs_s[n + 256] = x[b * XDIM + AH + 256 + n];
    __syncthreads();
    float acc = b1[n] + t[b] * W1[2 * HIDDEN + n];
    const float* w1o = W1 + 3 * HIDDEN + n;
    #pragma unroll 8
    for (int k = 0; k < OBS_DIM; ++k)
        acc = fmaf(obs_s[k], w1o[k * HIDDEN], acc);
    base[b * HIDDEN + n] = acc;
}

// ---------------------------------------------------------------------------
// Fused: h1 = silu(base + a*W1[0]) -> [WMMA f16 GEMM vs W2] -> silu(+b2)
//        -> dot with W3[:,0] -> out[:, :128]; zeros for out[:, 128:640]
// One workgroup (8 waves) per batch element. Each wave owns 2 N-tiles and
// keeps all 16 B tiles in VGPRs across the whole M loop. A tiles are staged
// into a distinct av[8] register array so all 16 ds_load_b128 issue before a
// single s_wait_dscnt, then 16 WMMAs run back-to-back.
// ---------------------------------------------------------------------------
__global__ void __launch_bounds__(256) fused_kernel(
    const float* __restrict__ x,    const float* __restrict__ W1,
    const float* __restrict__ base, const f16_t* __restrict__ w2t,
    const float* __restrict__ b2,   const float* __restrict__ W3,
    const float* __restrict__ b3,   float* __restrict__ out)
{
    __shared__ float act_s[AH];
    __shared__ float vel_s[AH];
    __shared__ __align__(16) f16_t h1s[64 * H1PITCH];   // 33.8 KB, two passes

    const int b    = blockIdx.x;
    const int tid  = threadIdx.x;
    const int lane = tid & 31;
    const int wave = tid >> 5;
    const int li   = lane & 15;   // column within tile / A row within tile
    const int hh   = lane >> 4;   // lane half -> K sub-pattern select

    if (tid < AH) { act_s[tid] = x[b * XDIM + tid]; vel_s[tid] = 0.0f; }

    // Per-thread layer-1 constants for channel k = tid
    const float basek = base[b * HIDDEN + tid];
    const float w1ak  = W1[tid];                        // W1 row 0 (action weight)

    // --- Load register-resident B tiles (v16h per 32-K chunk, per N-tile) ---
    // B (32x16 f16), this code's k order (same as A): lane L holds column n = L&15; element i of v16h
    // maps to K = 16*(L>>4) + i  =>  contiguous 16 halves of W2T column n.
    v16h  Breg[2][8];
    float biasn[2], w3n[2];
    #pragma unroll
    for (int tt = 0; tt < 2; ++tt) {
        const int n = (2 * wave + tt) * 16 + li;
        biasn[tt] = b2[n];
        w3n[tt]   = W3[2 * n];                          // W3[n, 0]
        const f16_t* colp = w2t + n * HIDDEN + hh * 16;
        #pragma unroll
        for (int kb = 0; kb < 8; ++kb) {
            v16h bv;
            ((v8h*)&bv)[0] = *(const v8h*)(colp + kb * 32);
            ((v8h*)&bv)[1] = *(const v8h*)(colp + kb * 32 + 8);
            Breg[tt][kb] = bv;
        }
    }

    for (int pass = 0; pass < 2; ++pass) {
        __syncthreads();                                // h1s reuse + act_s ready
        // ---- Layer 1 (rank-1 update + silu), rows j = pass*64 .. +63 ----
        #pragma unroll 4
        for (int j0 = 0; j0 < 64; ++j0) {
            const float a = act_s[pass * 64 + j0];
            const float v = fmaf(a, w1ak, basek);
            h1s[j0 * H1PITCH + tid] = (f16_t)silu_f(v);
        }
        __syncthreads();

        // ---- Layer 2: WMMA f32 += f16 x f16 ----
        for (int mtl = 0; mtl < 4; ++mtl) {
            // A (16x32 f16), this code's k order (same as B): lane holds row m = L&15; element i maps to
            // K = 16*(L>>4)+i.
            const f16_t* rowp = h1s + (mtl * 16 + li) * H1PITCH + hh * 16;

            // Stage the full 16x256 A strip for this M-tile (8 chunks, 64 VGPRs)
            v16h av[8];
            #pragma unroll
            for (int kb = 0; kb < 8; ++kb) {
                ((v8h*)&av[kb])[0] = *(const v8h*)(rowp + kb * 32);
                ((v8h*)&av[kb])[1] = *(const v8h*)(rowp + kb * 32 + 8);
            }

            v8f acc0 = {};
            v8f acc1 = {};
            #pragma unroll
            for (int kb = 0; kb < 8; ++kb) {
                acc0 = __builtin_amdgcn_wmma_f32_16x16x32_f16(
                    false, av[kb], false, Breg[0][kb], (short)0, acc0, false, false);
                acc1 = __builtin_amdgcn_wmma_f32_16x16x32_f16(
                    false, av[kb], false, Breg[1][kb], (short)0, acc1, false, false);
            }

            // ---- Epilogue: silu(h2 + b2) dotted with W3[:,0] ----
            // C/D layout: VGPR r holds row r + 8*(L>>4); column = L&15.
            float part[8];
            #pragma unroll
            for (int r = 0; r < 8; ++r) {
                const float h2a = silu_f(acc0[r] + biasn[0]) * w3n[0];
                const float h2b = silu_f(acc1[r] + biasn[1]) * w3n[1];
                part[r] = h2a + h2b;
            }
            #pragma unroll
            for (int r = 0; r < 8; ++r) {
                float v = part[r];
                v += __shfl_xor(v, 1, 16);
                v += __shfl_xor(v, 2, 16);
                v += __shfl_xor(v, 4, 16);
                v += __shfl_xor(v, 8, 16);
                if (li == 0) {
                    const int row = pass * 64 + mtl * 16 + hh * 8 + r;
                    atomicAdd(&vel_s[row], v);
                }
            }
        }
    }

    __syncthreads();
    if (tid < AH) out[b * XDIM + tid] = vel_s[tid] + b3[0];
    for (int i = tid; i < OBS_DIM; i += 256) out[b * XDIM + AH + i] = 0.0f;
}

extern "C" void kernel_launch(void* const* d_in, const int* in_sizes, int n_in,
                              void* d_out, int out_size, void* d_ws, size_t ws_size,
                              hipStream_t stream) {
    (void)in_sizes; (void)n_in; (void)out_size; (void)ws_size;
    const float* t  = (const float*)d_in[0];
    const float* x  = (const float*)d_in[1];
    const float* W1 = (const float*)d_in[2];
    const float* b1 = (const float*)d_in[3];
    const float* W2 = (const float*)d_in[4];
    const float* b2 = (const float*)d_in[5];
    const float* W3 = (const float*)d_in[6];
    const float* b3 = (const float*)d_in[7];
    float* out = (float*)d_out;

    float* base = (float*)d_ws;                                   // 512*256 f32
    f16_t* w2t  = (f16_t*)((char*)d_ws + BATCH * HIDDEN * sizeof(float)); // 256*256 f16

    w2t_kernel <<<HIDDEN, HIDDEN, 0, stream>>>(W2, w2t);
    base_kernel<<<BATCH,  HIDDEN, 0, stream>>>(t, x, W1, b1, base);
    fused_kernel<<<BATCH, 256, 0, stream>>>(x, W1, base, w2t, b2, W3, b3, out);
}
